// GCN_5763846111796
// MI455X (gfx1250) — matrix operand order repaired and checked by reading, not compiled, not run on hardware
//
#include <hip/hip_runtime.h>
#include <hip/hip_bf16.h>
#include <cstdint>

typedef __attribute__((ext_vector_type(16))) _Float16 v16h;
typedef __attribute__((ext_vector_type(8)))  float    v8f;

#define N_NODES     50000
#define N_EDGES     1600000
#define D_FEAT      128
#define H1C         64
#define H2C         128
#define NUM_CLASSES 40
#define NCP         48   // classes padded to a multiple of 16 for WMMA tiles

// ---------------------------------------------------------------------------
// Weight conversion f32 -> f16, TRANSPOSED: dst[n*K + k] = src[k*N + n]
// (rows n >= N are zero padding). Transposed layout makes each WMMA
// B-fragment a contiguous 32-byte run -> 2x global_load_b128 per fragment.
// ---------------------------------------------------------------------------
__global__ void __launch_bounds__(256)
cvt_weights_t(const float* __restrict__ src, _Float16* __restrict__ dst,
              int K, int N, int NP) {
    int i = blockIdx.x * 256 + threadIdx.x;
    if (i >= NP * K) return;
    int n = i / K, k = i - n * K;
    dst[i] = (n < N) ? (_Float16)src[k * N + n] : (_Float16)0.0f;
}

// ---------------------------------------------------------------------------
// Degree accumulation + dinv = rsqrt(deg + 1)
// ---------------------------------------------------------------------------
__global__ void __launch_bounds__(256)
deg_accum(const long long* __restrict__ ei, float* __restrict__ deg) {
    int e = blockIdx.x * 256 + threadIdx.x;
    if (e >= N_EDGES) return;
    atomicAdd(&deg[(int)ei[N_EDGES + e]], 1.0f);
}

__global__ void __launch_bounds__(256)
make_dinv(float* __restrict__ deg) {
    int i = blockIdx.x * 256 + threadIdx.x;
    if (i < N_NODES) deg[i] = rsqrtf(deg[i] + 1.0f);
}

// ---------------------------------------------------------------------------
// WMMA GEMM: C[M x NT*16] = A[MxK](f32) * Bt^T  (Bt is [NT*16 x K] f16)
// Each wave owns one 16-row stripe and all NT column tiles: the A fragment
// is loaded once per K-chunk and reused for NT back-to-back independent
// v_wmma_f32_16x16x32_f16 ops (no D->A/B RAW hazards, latency hidden).
// C/D layout per ISA 7.12.2; A/B both use contiguous k = 16*g + s, not the ISA table order.
// ---------------------------------------------------------------------------
template<int NT>
__global__ void __launch_bounds__(256)
wmma_gemm_t(const float* __restrict__ A, int lda,
            const _Float16* __restrict__ Bt,   // ld = K, rows = NT*16
            float* __restrict__ C, int ldc,
            const float* __restrict__ bias,
            int M, int K) {
    const int lane = threadIdx.x & 31;
    const int mt   = blockIdx.x * 8 + (threadIdx.x >> 5);
    if (mt >= (M >> 4)) return;             // wave-uniform: EXEC all-1s for WMMA

    const int m0 = mt << 4;
    const int g  = lane >> 4;               // half-wave group (0 or 1)
    const int l  = lane & 15;

    v8f acc[NT];
#pragma unroll
    for (int t = 0; t < NT; ++t) acc[t] = (v8f){};

    // A: lane holds row m0+l; group 0 covers K {0..15}, group 1 {16..31}
    const float*    Arow = A + (size_t)(m0 + l) * lda + (g << 4);
    // B fragment half h is K = k0 + g*16 + h for column n = t*16 + l
    const _Float16* Brow = Bt + (size_t)l * K + (g << 4);

    for (int k0 = 0; k0 < K; k0 += 32) {
        const float* ap = Arow + k0;
        float4 a0 = *(const float4*)(ap + 0);
        float4 a1 = *(const float4*)(ap + 4);
        float4 a2 = *(const float4*)(ap + 8);
        float4 a3 = *(const float4*)(ap + 12);
        __builtin_prefetch(ap + 32, 0, 3);  // keep near (WGP) scope
        v16h a;
        a[0]  = (_Float16)a0.x; a[1]  = (_Float16)a0.y; a[2]  = (_Float16)a0.z; a[3]  = (_Float16)a0.w;
        a[4]  = (_Float16)a1.x; a[5]  = (_Float16)a1.y; a[6]  = (_Float16)a1.z; a[7]  = (_Float16)a1.w;
        a[8]  = (_Float16)a2.x; a[9]  = (_Float16)a2.y; a[10] = (_Float16)a2.z; a[11] = (_Float16)a2.w;
        a[12] = (_Float16)a3.x; a[13] = (_Float16)a3.y; a[14] = (_Float16)a3.z; a[15] = (_Float16)a3.w;

#pragma unroll
        for (int t = 0; t < NT; ++t) {
            // contiguous 32B -> 2x global_load_b128
            v16h b = *(const v16h*)(Brow + (size_t)(t << 4) * K + k0);
            acc[t] = __builtin_amdgcn_wmma_f32_16x16x32_f16(
                /*neg_a=*/false, a, /*neg_b=*/false, b,
                /*c_mod=*/(short)0, acc[t], /*reuse_a=*/false, /*reuse_b=*/false);
        }
    }

#pragma unroll
    for (int t = 0; t < NT; ++t) {
        const int n = (t << 4) + l;
        if (n < ldc) {                      // mask stores for the padded-class tile
            float bv = bias ? bias[n] : 0.0f;
#pragma unroll
            for (int r = 0; r < 8; ++r)     // C/D layout: M = m0 + r + 8*g
                C[(size_t)(m0 + r + (g << 3)) * ldc + n] = acc[t][r] + bv;
        }
    }
}

// ---------------------------------------------------------------------------
// Edge aggregation: agg[dst] += h[src] * dinv[src]*dinv[dst]
// One wave per edge. The wave index is forced scalar (readfirstlane) so the
// edge endpoints and dinv values lower to SMEM loads (1x per edge, not 32x),
// and each lane gathers 2/4 consecutive channels with one b64/b128 load.
// Atomics land in L2 (agg fits in the 192MB L2).
// ---------------------------------------------------------------------------
template<int CSH>   // log2(channels): 6 or 7
__global__ void __launch_bounds__(256)
scatter_edges(const long long* __restrict__ ei,
              const float* __restrict__ h, const float* __restrict__ dinv,
              float* __restrict__ agg) {
    const int VEC = (1 << CSH) / 32;        // 2 (C=64) or 4 (C=128)
    unsigned w = (unsigned)__builtin_amdgcn_readfirstlane(
        blockIdx.x * 8u + (threadIdx.x >> 5));
    if (w >= N_EDGES) return;               // wave-uniform
    const int lane = threadIdx.x & 31;
    const int s = (int)ei[w];               // scalar loads (uniform address)
    const int d = (int)ei[N_EDGES + w];
    const float nm = dinv[s] * dinv[d];
    const float* hp = h + (((size_t)s << CSH) + lane * VEC);
    float*       ap = agg + (((size_t)d << CSH) + lane * VEC);
    if (VEC == 2) {
        float2 v = *(const float2*)hp;
        atomicAdd(ap + 0, v.x * nm);
        atomicAdd(ap + 1, v.y * nm);
    } else {
        float4 v = *(const float4*)hp;
        atomicAdd(ap + 0, v.x * nm);
        atomicAdd(ap + 1, v.y * nm);
        atomicAdd(ap + 2, v.z * nm);
        atomicAdd(ap + 3, v.w * nm);
    }
}

// ---------------------------------------------------------------------------
// Epilogue: h = dropout(tanh(agg + h * dinv^2 + bias))   (in-place on h)
// float4-vectorized: 4 channels per thread.
// ---------------------------------------------------------------------------
__device__ __forceinline__ unsigned hash32(unsigned x) {
    x ^= x >> 16; x *= 0x7feb352dU;
    x ^= x >> 15; x *= 0x846ca68bU;
    x ^= x >> 16; return x;
}

template<int CSH>
__global__ void __launch_bounds__(256)
finalize_act(float* __restrict__ h, const float* __restrict__ agg,
             const float* __restrict__ dinv, const float* __restrict__ bias,
             unsigned seed) {
    unsigned q = blockIdx.x * 256u + threadIdx.x;          // quad index
    unsigned total = ((unsigned)N_NODES << CSH) >> 2;
    if (q >= total) return;
    unsigned i = q << 2;
    unsigned n = i >> CSH;
    unsigned c = i & ((1u << CSH) - 1u);
    float di = dinv[n];
    float d2 = di * di;
    float4 hv = *(const float4*)(h + i);
    float4 av = *(const float4*)(agg + i);
    float4 bv = *(const float4*)(bias + c);
    float vi[4] = { av.x + hv.x * d2 + bv.x,
                    av.y + hv.y * d2 + bv.y,
                    av.z + hv.z * d2 + bv.z,
                    av.w + hv.w * d2 + bv.w };
    float vo[4];
#pragma unroll
    for (int j = 0; j < 4; ++j) {
        float v = tanhf(vi[j]);
        unsigned r = hash32((i + j) ^ seed);
        // keep with p=0.7: compare 24-bit hash against 0.7 * 2^24
        vo[j] = ((r & 0xFFFFFFu) < 11744051u) ? v * (1.0f / 0.7f) : 0.0f;
    }
    *(float4*)(h + i) = make_float4(vo[0], vo[1], vo[2], vo[3]);
}

// ---------------------------------------------------------------------------
extern "C" void kernel_launch(void* const* d_in, const int* in_sizes, int n_in,
                              void* d_out, int out_size, void* d_ws, size_t ws_size,
                              hipStream_t stream) {
    (void)in_sizes; (void)n_in; (void)out_size; (void)ws_size;

    const float*     x  = (const float*)d_in[0];
    const long long* ei = (const long long*)d_in[1];   // int64 [2, E]
    const float*     W1 = (const float*)d_in[2];
    const float*     b1 = (const float*)d_in[3];
    const float*     W2 = (const float*)d_in[4];
    const float*     b2 = (const float*)d_in[5];
    const float*     Wl = (const float*)d_in[6];
    const float*     bl = (const float*)d_in[7];
    float*           out = (float*)d_out;

    char* ws = (char*)d_ws;
    size_t off = 0;
    auto carve = [&](size_t bytes) -> char* {
        char* p = ws + off;
        off = (off + bytes + 255) & ~(size_t)255;
        return p;
    };
    float*    dinv = (float*)carve((size_t)N_NODES * 4);
    _Float16* W1t  = (_Float16*)carve((size_t)H1C * D_FEAT * 2);  // [64 x 128]
    _Float16* W2t  = (_Float16*)carve((size_t)H2C * H1C * 2);     // [128 x 64]
    _Float16* Wlt  = (_Float16*)carve((size_t)NCP * H2C * 2);     // [48 x 128]
    float*    bufA = (float*)carve((size_t)N_NODES * H2C * 4);    // 25.6 MB
    float*    bufB = (float*)carve((size_t)N_NODES * H2C * 4);    // 25.6 MB

    // --- weights to transposed f16 (Wl padded 40 -> 48 rows of zeros) ---
    cvt_weights_t<<<(H1C * D_FEAT + 255) / 256, 256, 0, stream>>>(W1, W1t, D_FEAT, H1C, H1C);
    cvt_weights_t<<<(H2C * H1C + 255) / 256, 256, 0, stream>>>(W2, W2t, H1C, H2C, H2C);
    cvt_weights_t<<<(NCP * H2C + 255) / 256, 256, 0, stream>>>(Wl, Wlt, H2C, NUM_CLASSES, NCP);

    // --- degrees / dinv ---
    hipMemsetAsync(dinv, 0, (size_t)N_NODES * 4, stream);
    deg_accum<<<(N_EDGES + 255) / 256, 256, 0, stream>>>(ei, dinv);
    make_dinv<<<(N_NODES + 255) / 256, 256, 0, stream>>>(dinv);

    const int rowTiles   = N_NODES / 16;        // 3125
    const int gemmBlocks = (rowTiles + 7) / 8;  // 8 waves per block
    const int edgeBlocks = (N_EDGES + 7) / 8;   // one wave per edge

    // --- layer 1: h = x @ W1 -> bufA ---
    wmma_gemm_t<4><<<gemmBlocks, 256, 0, stream>>>(
        x, D_FEAT, W1t, bufA, H1C, nullptr, N_NODES, D_FEAT);
    hipMemsetAsync(bufB, 0, (size_t)N_NODES * H1C * 4, stream);
    scatter_edges<6><<<edgeBlocks, 256, 0, stream>>>(ei, bufA, dinv, bufB);
    finalize_act<6><<<((unsigned)N_NODES * H1C / 4 + 255) / 256, 256, 0, stream>>>(
        bufA, bufB, dinv, b1, 0x9E3779B9u);

    // --- layer 2: h2 = h1 @ W2 -> bufB ---
    wmma_gemm_t<8><<<gemmBlocks, 256, 0, stream>>>(
        bufA, H1C, W2t, bufB, H2C, nullptr, N_NODES, H1C);
    hipMemsetAsync(bufA, 0, (size_t)N_NODES * H2C * 4, stream);
    scatter_edges<7><<<edgeBlocks, 256, 0, stream>>>(ei, bufB, dinv, bufA);
    finalize_act<7><<<((unsigned)N_NODES * H2C / 4 + 255) / 256, 256, 0, stream>>>(
        bufB, bufA, dinv, b2, 0x85EBCA6Bu);

    // --- output layer: out = h2 @ Wl + bl (stores masked to 40 classes) ---
    wmma_gemm_t<3><<<gemmBlocks, 256, 0, stream>>>(
        bufB, H2C, Wlt, out, NUM_CLASSES, bl, N_NODES, H2C);
}
